// TransformerCritic_81621558493518
// MI455X (gfx1250) — matrix operand order repaired and checked by reading, not compiled, not run on hardware
//
#include <hip/hip_runtime.h>
#include <math.h>

// ---------------------------------------------------------------------------
// TransformerCritic for MI455X (gfx1250, wave32, WMMA f32_16x16x32_f16)
// ---------------------------------------------------------------------------
// B=4096, N=8, A=5, D=128.  oap[b,i,j] only depends on j (except diagonal),
// so the 47->128->128 MLP runs on 2*B*N rows instead of B*N*N (4x FLOP cut).
// node_features = S[b,a,:] + w[b,a,i]*(av_p - av_a), S = w @ av_a.
// All GEMMs run as f16 WMMA with f32 accumulate.  GEMM waves own a 32(M) x
// 128(N) C stripe: per k-step, batch-load 1+1 A frags + 8 B frags, then issue
// 16 back-to-back WMMAs (B fragments reused across the two M tiles).
// ---------------------------------------------------------------------------

typedef __attribute__((ext_vector_type(16))) _Float16 v16h;
typedef __attribute__((ext_vector_type(8)))  _Float16 v8h;
typedef __attribute__((ext_vector_type(8)))  float    v8f;

#define BQ 4096
#define BN 32768            // B*N agents
#define R2 65536            // 2*BN rows (action rows then policy rows)
#define VAL_COUNT 262144    // B*N*N value outputs (== weight outputs)

// ---------------- fragment loaders (per CDNA5 ISA 7.12.2 layouts) ----------
// A (16x32 f16): lane holds row r=lane&15; h=lane>>4 selects interleaved
// 8-halfword chunks at k = kb+h*8 .. +7 and kb+16+h*8 .. +7.
__device__ __forceinline__ v16h load_a_frag(const _Float16* row, int kb, int h) {
  v8h lo = *(const v8h*)(row + kb + h * 8);
  v8h hi = *(const v8h*)(row + kb + 16 + h * 8);
  v16h r;
#pragma unroll
  for (int i = 0; i < 8; ++i) { r[i] = lo[i]; r[8 + i] = hi[i]; }
  return r;
}
// B (32x16 f16): lane holds col n=lane&15; h=lane>>4 selects interleaved
// 8-halfword chunks at k = kb+h*8 .. +7 and kb+16+h*8 .. +7 (from transposed weights Wt[n][k]).
__device__ __forceinline__ v16h load_b_frag(const _Float16* col, int kb, int h) {
  v8h lo = *(const v8h*)(col + kb + h * 8);
  v8h hi = *(const v8h*)(col + kb + 16 + h * 8);
  v16h r;
#pragma unroll
  for (int i = 0; i < 8; ++i) { r[i] = lo[i]; r[8 + i] = hi[i]; }
  return r;
}

// ---------------- K0: weight transpose + f32->f16 with K zero-padding ------
// src: [K][N] row-major f32.  dst: [N][Kpad] f16 (Wt[n][k] = W[k][n]).
__global__ void transpose_pad_kernel(const float* __restrict__ src,
                                     _Float16* __restrict__ dst,
                                     int K, int N, int Kpad) {
  int i = blockIdx.x * blockDim.x + threadIdx.x;
  if (i >= N * Kpad) return;
  int n = i / Kpad, k = i - n * Kpad;
  dst[i] = (_Float16)((k < K) ? src[k * N + n] : 0.0f);
}

// ---------------- K1: CNN encoder + build f16 GEMM rows --------------------
// One thread per agent.  conv2 output is 1x1 so conv2 is a full dot over
// the 16x3x3 conv1 activation -> accumulate acc2[32] on the fly.
__global__ void __launch_bounds__(256) cnn_rows_kernel(
    const float* __restrict__ st,       // [BN,3,5,5]
    const float* __restrict__ pos,      // [BN,2]
    const float* __restrict__ onehot,   // [BN,8]
    const float* __restrict__ pol,      // [BN,5]
    const float* __restrict__ act,      // [BN,5]
    const float* __restrict__ c1w, const float* __restrict__ c1b,
    const float* __restrict__ c2w, const float* __restrict__ c2b,
    _Float16* __restrict__ A_se,        // [BN,64]   (states padded 42->64)
    _Float16* __restrict__ rows_ap) {   // [2*BN,64] (states+actions | states+policies)
  __shared__ float s1w[432], s1b[16], s2w[4608], s2b[32];
  int tid = threadIdx.x;
  for (int i = tid; i < 432; i += 256) s1w[i] = c1w[i];
  for (int i = tid; i < 4608; i += 256) s2w[i] = c2w[i];
  if (tid < 16) s1b[tid] = c1b[tid];
  if (tid < 32) s2b[tid] = c2b[tid];
  __syncthreads();

  int g = blockIdx.x * 256 + tid;       // agent index < BN
  float in[75];
#pragma unroll
  for (int i = 0; i < 75; ++i) in[i] = st[(size_t)g * 75 + i];

  float acc2[32];
#pragma unroll
  for (int oc = 0; oc < 32; ++oc) acc2[oc] = s2b[oc];

  for (int c1 = 0; c1 < 16; ++c1) {
    int base1 = c1 * 27;
#pragma unroll
    for (int y = 0; y < 3; ++y) {
#pragma unroll
      for (int x = 0; x < 3; ++x) {
        float v = s1b[c1];
#pragma unroll
        for (int ci = 0; ci < 3; ++ci)
#pragma unroll
          for (int ky = 0; ky < 3; ++ky)
#pragma unroll
            for (int kx = 0; kx < 3; ++kx)
              v += in[ci * 25 + (y + ky) * 5 + (x + kx)] *
                   s1w[base1 + ci * 9 + ky * 3 + kx];
        v = (v >= 0.0f) ? v : 0.01f * v;   // leaky relu
        int base2 = c1 * 9 + y * 3 + x;
#pragma unroll
        for (int oc = 0; oc < 32; ++oc) acc2[oc] += v * s2w[oc * 144 + base2];
      }
    }
  }

  float sarr[42];
#pragma unroll
  for (int oc = 0; oc < 32; ++oc) {
    float v = acc2[oc];
    sarr[oc] = (v >= 0.0f) ? v : 0.01f * v;
  }
  sarr[32] = pos[(size_t)g * 2 + 0];
  sarr[33] = pos[(size_t)g * 2 + 1];
#pragma unroll
  for (int j = 0; j < 8; ++j) sarr[34 + j] = onehot[(size_t)g * 8 + j];

  _Float16* r0 = A_se + (size_t)g * 64;
  _Float16* ra = rows_ap + (size_t)g * 64;
  _Float16* rp = rows_ap + (size_t)(BN + g) * 64;
#pragma unroll
  for (int i = 0; i < 42; ++i) {
    _Float16 hv = (_Float16)sarr[i];
    r0[i] = hv; ra[i] = hv; rp[i] = hv;
  }
#pragma unroll
  for (int i = 42; i < 64; ++i) r0[i] = (_Float16)0.0f;
#pragma unroll
  for (int j = 0; j < 5; ++j) {
    ra[42 + j] = (_Float16)act[(size_t)g * 5 + j];
    rp[42 + j] = (_Float16)pol[(size_t)g * 5 + j];
  }
#pragma unroll
  for (int i = 47; i < 64; ++i) { ra[i] = (_Float16)0.0f; rp[i] = (_Float16)0.0f; }
}

// ---------------- K2: generic WMMA GEMM, N=128, out = act(A@W + b) ---------
// 256 threads = 8 waves; each wave owns a 32(M) x 128(N) C stripe
// (2 x 8 accumulator tiles; B fragments reused across the two M tiles).
// K = KSTEPS*32 (zero-padded).  Loads batched per k-step so the 8+8 WMMAs
// issue back-to-back after a single load clause.
template <int KSTEPS>
__global__ void __launch_bounds__(256) gemm128_kernel(
    const _Float16* __restrict__ A, int lda,
    const _Float16* __restrict__ Wt, int ldk,
    const float* __restrict__ bias,
    _Float16* __restrict__ out,
    int do_tanh) {
  int lane = threadIdx.x & 31;
  int wave = threadIdx.x >> 5;
  int h = lane >> 4, n0 = lane & 15;
  int mBase = (blockIdx.x * 8 + wave) * 32;
  const _Float16* arow0 = A + (size_t)(mBase + n0) * lda;
  const _Float16* arow1 = A + (size_t)(mBase + 16 + n0) * lda;
  const _Float16* cols[8];
#pragma unroll
  for (int t = 0; t < 8; ++t) cols[t] = Wt + (size_t)(16 * t + n0) * ldk;

  v8f acc0[8], acc1[8];
#pragma unroll
  for (int t = 0; t < 8; ++t)
#pragma unroll
    for (int i = 0; i < 8; ++i) { acc0[t][i] = 0.0f; acc1[t][i] = 0.0f; }

#pragma unroll
  for (int kk = 0; kk < KSTEPS; ++kk) {
    int kb = kk * 32;
    v16h a0 = load_a_frag(arow0, kb, h);
    v16h a1 = load_a_frag(arow1, kb, h);
    v16h bf[8];
#pragma unroll
    for (int t = 0; t < 8; ++t) bf[t] = load_b_frag(cols[t], kb, h);
#pragma unroll
    for (int t = 0; t < 8; ++t) {
      acc0[t] = __builtin_amdgcn_wmma_f32_16x16x32_f16(
          false, a0, false, bf[t], (short)0, acc0[t], false, false);
      acc1[t] = __builtin_amdgcn_wmma_f32_16x16x32_f16(
          false, a1, false, bf[t], (short)0, acc1[t], false, false);
    }
  }

#pragma unroll
  for (int t = 0; t < 8; ++t) {
    int n = 16 * t + n0;
    float bv = bias[n];
#pragma unroll
    for (int i = 0; i < 8; ++i) {
      float v0 = acc0[t][i] + bv;
      float v1 = acc1[t][i] + bv;
      if (do_tanh) { v0 = tanhf(v0); v1 = tanhf(v1); }
      out[(size_t)(mBase + i + 8 * h) * 128 + n] = (_Float16)v0;
      out[(size_t)(mBase + 16 + i + 8 * h) * 128 + n] = (_Float16)v1;
    }
  }
}

// ---------------- K3: attention scores + softmax (N=8, VALU) ---------------
__global__ void __launch_bounds__(128) attn_kernel(
    const _Float16* __restrict__ q, const _Float16* __restrict__ k,
    float* __restrict__ wout) {
  int t = blockIdx.x * 128 + threadIdx.x;   // (b,a) index < BN
  int b = t >> 3;
  const _Float16* qr = q + (size_t)t * 128;
  float sc[8];
  float mx = -1e30f;
#pragma unroll
  for (int j = 0; j < 8; ++j) {
    const _Float16* kr = k + ((size_t)(b * 8 + j)) * 128;
    float s = 0.0f;
    for (int d = 0; d < 128; ++d) s += (float)qr[d] * (float)kr[d];
    sc[j] = s * 0.088388347648318447f;      // 1/sqrt(128)
    mx = fmaxf(mx, sc[j]);
  }
  float sum = 0.0f;
#pragma unroll
  for (int j = 0; j < 8; ++j) { sc[j] = __expf(sc[j] - mx); sum += sc[j]; }
  float inv = 1.0f / sum;
#pragma unroll
  for (int j = 0; j < 8; ++j) wout[(size_t)t * 8 + j] = sc[j] * inv;
}

// ---------------- K4: fused node_features + f1 (WMMA) + f2 -----------------
// One block (4 waves) per batch b.  nf[64][128] built in LDS, then
// 64x128 @ 128x64 WMMA, tanh, dot f2_w via shfl_xor reduction.
__global__ void __launch_bounds__(128) final_kernel(
    const _Float16* __restrict__ av,      // [2*BN,128]: av_a rows, then av_p
    const float* __restrict__ wgt,        // [B,8,8] softmax weights
    const _Float16* __restrict__ f1wt,    // [64][128] transposed f16
    const float* __restrict__ f1b,        // [64]
    const float* __restrict__ f2w,        // [64]
    const float* __restrict__ f2b,        // [1]
    float* __restrict__ val) {            // [B,64]
  __shared__ float s_w[64];
  __shared__ float s_S[8 * 128];
  __shared__ __align__(16) _Float16 s_nf[64 * 128];

  int b = blockIdx.x, tid = threadIdx.x;
  if (tid < 64) s_w[tid] = wgt[(size_t)b * 64 + tid];
  __syncthreads();

  // S[a,d] = sum_j w[a,j] * av_a[b,j,d]
  for (int idx = tid; idx < 1024; idx += 128) {
    int a = idx >> 7, d = idx & 127;
    float s = 0.0f;
#pragma unroll
    for (int j = 0; j < 8; ++j)
      s += s_w[a * 8 + j] * (float)av[((size_t)(b * 8 + j)) * 128 + d];
    s_S[idx] = s;
  }
  __syncthreads();

  // nf[a*8+i, d] = S[a,d] + w[a,i] * (av_p[b,i,d] - av_a[b,i,d])
  const size_t PB = (size_t)BN * 128;
  for (int idx = tid; idx < 8192; idx += 128) {
    int r = idx >> 7, d = idx & 127;
    int a = r >> 3, i = r & 7;
    float fa = (float)av[((size_t)(b * 8 + i)) * 128 + d];
    float fp = (float)av[PB + ((size_t)(b * 8 + i)) * 128 + d];
    s_nf[idx] = (_Float16)(s_S[a * 128 + d] + s_w[a * 8 + i] * (fp - fa));
  }
  __syncthreads();

  // WMMA: 16 rows per wave, 4 N-tiles (64 cols), 4 k-steps (K=128)
  int lane = tid & 31, wave = tid >> 5;
  int h = lane >> 4, n0 = lane & 15;
  const _Float16* arow = s_nf + (size_t)(wave * 16 + n0) * 128;
  const _Float16* cols[4];
#pragma unroll
  for (int t = 0; t < 4; ++t) cols[t] = f1wt + (size_t)(16 * t + n0) * 128;

  v8f acc[4];
#pragma unroll
  for (int t = 0; t < 4; ++t)
#pragma unroll
    for (int i = 0; i < 8; ++i) acc[t][i] = 0.0f;

#pragma unroll
  for (int kk = 0; kk < 4; ++kk) {
    int kb = kk * 32;
    v16h a = load_a_frag(arow, kb, h);
    v16h bf[4];
#pragma unroll
    for (int t = 0; t < 4; ++t) bf[t] = load_b_frag(cols[t], kb, h);
#pragma unroll
    for (int t = 0; t < 4; ++t)
      acc[t] = __builtin_amdgcn_wmma_f32_16x16x32_f16(
          false, a, false, bf[t], (short)0, acc[t], false, false);
  }

  // v = tanh(acc + f1_b); partial = v * f2_w; reduce over 16 cols per half.
  float p[8] = {0, 0, 0, 0, 0, 0, 0, 0};
#pragma unroll
  for (int t = 0; t < 4; ++t) {
    int n = 16 * t + n0;
    float bv = f1b[n];
    float fw = f2w[n];
#pragma unroll
    for (int i = 0; i < 8; ++i) p[i] += tanhf(acc[t][i] + bv) * fw;
  }
#pragma unroll
  for (int i = 0; i < 8; ++i)
#pragma unroll
    for (int m = 1; m < 16; m <<= 1) p[i] += __shfl_xor(p[i], m, 32);

  if (n0 == 0) {
    float fb = f2b[0];
#pragma unroll
    for (int i = 0; i < 8; ++i)
      val[(size_t)b * 64 + wave * 16 + i + 8 * h] = p[i] + fb;
  }
}

// ---------------------------------------------------------------------------
extern "C" void kernel_launch(void* const* d_in, const int* in_sizes, int n_in,
                              void* d_out, int out_size, void* d_ws, size_t ws_size,
                              hipStream_t stream) {
  (void)in_sizes; (void)n_in; (void)out_size; (void)ws_size;

  const float* agent_states = (const float*)d_in[0];
  const float* positions    = (const float*)d_in[1];
  const float* onehot       = (const float*)d_in[2];
  const float* policies     = (const float*)d_in[3];
  const float* actions      = (const float*)d_in[4];
  const float* conv1_w = (const float*)d_in[5];
  const float* conv1_b = (const float*)d_in[6];
  const float* conv2_w = (const float*)d_in[7];
  const float* conv2_b = (const float*)d_in[8];
  const float* se_w  = (const float*)d_in[9];
  const float* se_b  = (const float*)d_in[10];
  const float* key_w = (const float*)d_in[11];
  const float* key_b = (const float*)d_in[12];
  const float* query_w = (const float*)d_in[13];
  const float* query_b = (const float*)d_in[14];
  const float* sap_w = (const float*)d_in[15];
  const float* sap_b = (const float*)d_in[16];
  const float* av_w  = (const float*)d_in[17];
  const float* av_b  = (const float*)d_in[18];
  const float* f1_w  = (const float*)d_in[19];
  const float* f1_b  = (const float*)d_in[20];
  const float* f2_w  = (const float*)d_in[21];
  const float* f2_b  = (const float*)d_in[22];

  float* out = (float*)d_out;
  float* value_out  = out;               // [B,N,N,1] = 262144 floats
  float* weight_out = out + VAL_COUNT;   // [B,N,N]   = 262144 floats

  // ---- workspace layout (256B aligned) ----
  char* ws = (char*)d_ws;
  size_t o = 0;
  auto take = [&](size_t bytes) -> void* {
    void* p = ws + o;
    o += (bytes + 255) & ~(size_t)255;
    return p;
  };
  _Float16* se_wt    = (_Float16*)take((size_t)128 * 64 * 2);
  _Float16* key_wt   = (_Float16*)take((size_t)128 * 128 * 2);
  _Float16* query_wt = (_Float16*)take((size_t)128 * 128 * 2);
  _Float16* sap_wt   = (_Float16*)take((size_t)128 * 64 * 2);
  _Float16* av_wt    = (_Float16*)take((size_t)128 * 128 * 2);
  _Float16* f1_wt    = (_Float16*)take((size_t)64 * 128 * 2);
  _Float16* A_se     = (_Float16*)take((size_t)BN * 64 * 2);
  _Float16* rows_ap  = (_Float16*)take((size_t)R2 * 64 * 2);
  _Float16* se_emb   = (_Float16*)take((size_t)BN * 128 * 2);
  _Float16* qbuf     = (_Float16*)take((size_t)BN * 128 * 2);
  _Float16* kbuf     = (_Float16*)take((size_t)BN * 128 * 2);
  _Float16* oe       = (_Float16*)take((size_t)R2 * 128 * 2);
  _Float16* avb      = (_Float16*)take((size_t)R2 * 128 * 2);

  // ---- K0: weight prep (transpose, f16, K zero-pad) ----
  transpose_pad_kernel<<<(128 * 64 + 255) / 256, 256, 0, stream>>>(se_w, se_wt, 42, 128, 64);
  transpose_pad_kernel<<<(128 * 128 + 255) / 256, 256, 0, stream>>>(key_w, key_wt, 128, 128, 128);
  transpose_pad_kernel<<<(128 * 128 + 255) / 256, 256, 0, stream>>>(query_w, query_wt, 128, 128, 128);
  transpose_pad_kernel<<<(128 * 64 + 255) / 256, 256, 0, stream>>>(sap_w, sap_wt, 47, 128, 64);
  transpose_pad_kernel<<<(128 * 128 + 255) / 256, 256, 0, stream>>>(av_w, av_wt, 128, 128, 128);
  transpose_pad_kernel<<<(64 * 128 + 255) / 256, 256, 0, stream>>>(f1_w, f1_wt, 128, 64, 128);

  // ---- K1: CNN + GEMM-row construction ----
  cnn_rows_kernel<<<BN / 256, 256, 0, stream>>>(
      agent_states, positions, onehot, policies, actions,
      conv1_w, conv1_b, conv2_w, conv2_b, A_se, rows_ap);

  // ---- K2 x5: WMMA GEMMs (32 rows per wave, 8 waves per block) ----
  // states_embed = tanh(states @ se_w + se_b)          [32768 x 42 -> 128]
  gemm128_kernel<2><<<BN / 256, 256, 0, stream>>>(A_se, 64, se_wt, 64, se_b, se_emb, 1);
  // q / k projections                                   [32768 x 128 -> 128]
  gemm128_kernel<4><<<BN / 256, 256, 0, stream>>>(se_emb, 128, query_wt, 128, query_b, qbuf, 0);
  gemm128_kernel<4><<<BN / 256, 256, 0, stream>>>(se_emb, 128, key_wt, 128, key_b, kbuf, 0);
  // oap_embed = tanh(rows @ sap_w + sap_b)             [65536 x 47 -> 128]
  gemm128_kernel<2><<<R2 / 256, 256, 0, stream>>>(rows_ap, 64, sap_wt, 64, sap_b, oe, 1);
  // attn_vals = tanh(oap_embed @ av_w + av_b)          [65536 x 128 -> 128]
  gemm128_kernel<4><<<R2 / 256, 256, 0, stream>>>(oe, 128, av_wt, 128, av_b, avb, 1);

  // ---- K3: softmax attention weights (also output #2) ----
  attn_kernel<<<BN / 128, 128, 0, stream>>>(qbuf, kbuf, weight_out);

  // ---- K4: fused node_features + f1 (WMMA) + f2 -> value output ----
  final_kernel<<<BQ, 128, 0, stream>>>(avb, weight_out, f1_wt, f1_b, f2_w, f2_b, value_out);
}
